// DFineLQE_8839042695535
// MI455X (gfx1250) — compile-verified, matrix operand order checked by reading, not run on hardware
//
#include <hip/hip_runtime.h>

typedef _Float16 v16h __attribute__((ext_vector_type(16)));
typedef _Float16 v8h  __attribute__((ext_vector_type(8)));
typedef float    v8f  __attribute__((ext_vector_type(8)));
typedef float    v4f  __attribute__((ext_vector_type(4)));

#define NBINS     33      // MAX_NUM_BINS + 1
#define STAT_DIM  20
#define KPAD      32      // pad K 20 -> 32 for 16x16x32 f16 WMMA
#define HIDDEN    64
#define C_DIM     80
#define TOK_BLK   256     // tokens per workgroup (8 wave32)

__launch_bounds__(TOK_BLK)
__global__ void lqe_kernel(const float* __restrict__ scores,
                           const float* __restrict__ corners,
                           const float* __restrict__ W1,
                           const float* __restrict__ b1,
                           const float* __restrict__ W2,
                           const float* __restrict__ b2,
                           float* __restrict__ out)
{
    __shared__ __align__(16) _Float16 s_stat[TOK_BLK * KPAD];  // 16 KB
    __shared__ __align__(16) _Float16 s_w1[KPAD * HIDDEN];     // 4 KB (zero-padded rows 20..31)
    __shared__ float s_b1[HIDDEN];
    __shared__ float s_w2[HIDDEN];
    __shared__ float s_b2;
    __shared__ __align__(16) float s_h[TOK_BLK * HIDDEN];      // 64 KB

    const int tid   = threadIdx.x;
    const int token = blockIdx.x * TOK_BLK + tid;

    // ---------------- stage weights (one-time per block) ----------------
    for (int i = tid; i < KPAD * HIDDEN; i += TOK_BLK) {
        const int k = i >> 6, n = i & 63;
        const float w = (k < STAT_DIM) ? W1[k * HIDDEN + n] : 0.0f;
        s_w1[i] = (_Float16)w;
    }
    if (tid < HIDDEN) { s_b1[tid] = b1[tid]; s_w2[tid] = W2[tid]; }
    if (tid == 0)     { s_b2 = b2[0]; }

    // ---------------- phase 1: softmax + top-4 stats (thread-per-token) ----------------
    {
        const float* cp = corners + (size_t)token * (4 * NBINS);
        _Float16* sp = &s_stat[tid * KPAD];
        #pragma unroll
        for (int g = 0; g < 4; ++g) {
            float v[NBINS];
            #pragma unroll
            for (int i = 0; i < NBINS; ++i) v[i] = cp[g * NBINS + i];

            // branchless top-4 (descending) of logits; softmax is monotonic
            float t0 = -INFINITY, t1 = -INFINITY, t2 = -INFINITY, t3 = -INFINITY;
            #pragma unroll
            for (int i = 0; i < NBINS; ++i) {
                float x = v[i], y;
                y = fminf(t0, x); t0 = fmaxf(t0, x); x = y;
                y = fminf(t1, x); t1 = fmaxf(t1, x); x = y;
                y = fminf(t2, x); t2 = fmaxf(t2, x); x = y;
                t3 = fmaxf(t3, x);
            }
            float s = 0.0f;
            #pragma unroll
            for (int i = 0; i < NBINS; ++i) s += __expf(v[i] - t0);
            const float inv = 1.0f / s;
            const float p0 = inv;                     // exp(t0 - t0) == 1
            const float p1 = __expf(t1 - t0) * inv;
            const float p2 = __expf(t2 - t0) * inv;
            const float p3 = __expf(t3 - t0) * inv;
            const float mn = 0.25f * (p0 + p1 + p2 + p3);
            sp[g * 5 + 0] = (_Float16)p0;
            sp[g * 5 + 1] = (_Float16)p1;
            sp[g * 5 + 2] = (_Float16)p2;
            sp[g * 5 + 3] = (_Float16)p3;
            sp[g * 5 + 4] = (_Float16)mn;
        }
        #pragma unroll
        for (int k = STAT_DIM; k < KPAD; ++k) sp[k] = (_Float16)0.0f;
    }
    __syncthreads();

    // ---------------- phase 2: 20->64 layer via v_wmma_f32_16x16x32_f16 ----------------
    {
        const int lane = tid & 31;
        const int wave = tid >> 5;        // 0..7
        const int hh   = lane >> 4;       // half: 0 or 1
        const int l16  = lane & 15;

        // B tiles (W1): lane<16 holds col N=l16, K {0..7,16..23}; lane>=16 K {8..15,24..31}
        v16h Bt[4];
        #pragma unroll
        for (int t = 0; t < 4; ++t) {
            const int col = t * 16 + l16;
            #pragma unroll
            for (int i = 0; i < 8; ++i) {
                Bt[t][i]     = s_w1[(hh * 8 + i) * HIDDEN + col];
                Bt[t][i + 8] = s_w1[(16 + hh * 8 + i) * HIDDEN + col];
            }
        }

        #pragma unroll
        for (int tt = 0; tt < 2; ++tt) {
            const int tile   = wave * 2 + tt;          // 16 tiles of 16 tokens
            const int rowTok = tile * 16 + l16;
            // A: lane<16 row M=l16, K {0..7,16..23}; lane>=16 K {8..15,24..31}
            const _Float16* ap = &s_stat[rowTok * KPAD + hh * 8];
            const v8h lo = *(const v8h*)(ap);
            const v8h hi = *(const v8h*)(ap + 16);
            v16h A;
            #pragma unroll
            for (int i = 0; i < 8; ++i) { A[i] = lo[i]; A[i + 8] = hi[i]; }

            #pragma unroll
            for (int t = 0; t < 4; ++t) {
                const float bv = s_b1[t * 16 + l16];   // bias fused into accumulator
                v8f c = {bv, bv, bv, bv, bv, bv, bv, bv};
                v8f d = __builtin_amdgcn_wmma_f32_16x16x32_f16(
                    /*neg_a=*/false, A, /*neg_b=*/false, Bt[t],
                    /*c_mod=*/(short)0, c, /*reuse_a=*/false, /*reuse_b=*/false);
                #pragma unroll
                for (int r = 0; r < 8; ++r) {
                    const int m = r + hh * 8;          // D: VGPR r holds row r (lanes<16) / r+8
                    s_h[(tile * 16 + m) * HIDDEN + t * 16 + l16] = fmaxf(d[r], 0.0f);
                }
            }
        }
    }
    __syncthreads();

    // ---------------- phase 3: 64->1 dot + streaming add (thread-per-token) ----------------
    {
        float q = s_b2;
        const float* hp = &s_h[tid * HIDDEN];
        #pragma unroll
        for (int h = 0; h < HIDDEN; ++h) q += hp[h] * s_w2[h];

        const v4f* sc = (const v4f*)(scores + (size_t)token * C_DIM);
        v4f*       op = (v4f*)(out + (size_t)token * C_DIM);
        #pragma unroll
        for (int i = 0; i < C_DIM / 4; ++i) {
            v4f v = __builtin_nontemporal_load(sc + i);   // 612 MB stream >> 192 MB L2
            v += (v4f){q, q, q, q};
            __builtin_nontemporal_store(v, op + i);
        }
    }
}

extern "C" void kernel_launch(void* const* d_in, const int* in_sizes, int n_in,
                              void* d_out, int out_size, void* d_ws, size_t ws_size,
                              hipStream_t stream) {
    const float* scores  = (const float*)d_in[0];
    const float* corners = (const float*)d_in[1];
    const float* W1      = (const float*)d_in[2];
    const float* b1      = (const float*)d_in[3];
    const float* W2      = (const float*)d_in[4];
    const float* b2      = (const float*)d_in[5];
    float* out = (float*)d_out;

    const int n_tokens = in_sizes[1] / (4 * NBINS);   // 524288, divisible by TOK_BLK
    const int blocks   = n_tokens / TOK_BLK;          // 2048
    lqe_kernel<<<blocks, TOK_BLK, 0, stream>>>(scores, corners, W1, b1, W2, b2, out);
}
